// HeatmapGuidedPlacer_77429670412469
// MI455X (gfx1250) — matrix operand order repaired and checked by reading, not compiled, not run on hardware
//
#include <hip/hip_runtime.h>
#include <hip/hip_bf16.h>
#include <math.h>

typedef __attribute__((ext_vector_type(16))) __bf16 v16bf;
typedef __attribute__((ext_vector_type(8)))  __bf16 v8bf;
typedef __attribute__((ext_vector_type(8)))  float  v8f;
typedef __attribute__((ext_vector_type(4)))  unsigned int u32x4;
typedef __attribute__((ext_vector_type(8)))  int i32x8;
typedef __attribute__((ext_vector_type(4)))  int i32x4;

#define BATCH 16
#define HS 256
#define WSp 256
#define HO 1024
#define WOp 1024
#define NUM_CAND 5
#define TOPK 20
#define SCENE_DIM 512

// workspace byte offsets (all 64KB-aligned)
#define OFF_RESIZED 0UL                               // f32 [16,256,256]  (4 MB)
#define OFF_PW1 (OFF_RESIZED + 16UL*256*256*4)        // bf16 packed W1: 18*2*32*16
#define OFF_PW2 (OFF_PW1 + 65536UL)                   // bf16 packed W2: 9*32*16
#define OFF_C1  (OFF_PW2 + 65536UL)                   // bf16 [16,256,256,32ch] (64 MB)
#define OFF_C2  (OFF_C1 + 16UL*32*256*256*2)          // bf16 [16,256,256,16ch] (32 MB)
#define OFF_REF (OFF_C2 + 16UL*16*256*256*2)          // f32  [16,256,256]  (4 MB)

// ---------- TDM: 1-D contiguous block global -> LDS (CDNA5 Tensor Data Mover) ----
// D# group0: count=1 | lds_addr[63:32] | global_addr[120:64] | type=2 [127:126]
// D# group1: data_size=1B; tensor_dim0 = tile_dim0 = tensor_dim0_stride = nbytes
// This toolchain exposes the 6-arg builtin (extra i32x8 group, zero-filled).
__device__ __forceinline__ void tdm_load_1d(unsigned lds_off, const void* gptr,
                                            unsigned nbytes) {
    unsigned long long ga = (unsigned long long)(uintptr_t)gptr;
    u32x4 g0;
    g0[0] = 1u;                                            // count=1 (valid user D#)
    g0[1] = lds_off;                                       // lds_addr
    g0[2] = (unsigned)(ga & 0xffffffffu);                  // global_addr[31:0]
    g0[3] = (unsigned)((ga >> 32) & 0x1ffffffu) | (2u << 30); // addr[56:32] | type=2
    i32x8 g1;
    g1[0] = 0;                                             // wg_mask=0, data_size=1B
    g1[1] = (int)((nbytes & 0xffffu) << 16);               // tensor_dim0[15:0]
    g1[2] = (int)(((nbytes >> 16) & 0xffffu) | (1u << 16));// dim0[31:16], tensor_dim1=1
    g1[3] = (int)((nbytes & 0xffffu) << 16);               // tile_dim0 = nbytes
    g1[4] = 0;                                             // tile_dim1/2 unused
    g1[5] = (int)nbytes;                                   // tensor_dim0_stride lo
    g1[6] = 0;
    g1[7] = 0;
    i32x4 g2 = {0, 0, 0, 0};
    i32x4 g3 = {0, 0, 0, 0};
    i32x8 g4 = {0, 0, 0, 0, 0, 0, 0, 0};
    __builtin_amdgcn_tensor_load_to_lds(g0, g1, g2, g3, g4, 0);
}

// ---------- antialiased bilinear 1024 -> 256 (triangle kernel, scale 4) ----------
__global__ void k_downsample(const float* __restrict__ hm, float* __restrict__ resized) {
    int o = blockIdx.x * blockDim.x + threadIdx.x;
    if (o >= BATCH * HS * WSp) return;
    int b = o / (HS * WSp); int rem = o % (HS * WSp);
    int yo = rem / WSp, xo = rem % WSp;
    float cy = (yo + 0.5f) * 4.f - 0.5f;
    float cx = (xo + 0.5f) * 4.f - 0.5f;
    int ys = (int)ceilf(cy - 4.f), xs = (int)ceilf(cx - 4.f);
    float acc = 0.f, wsum = 0.f;
    for (int iy = 0; iy < 9; ++iy) {
        int y = ys + iy; if (y < 0 || y >= HO) continue;
        float wy = 1.f - fabsf((float)y - cy) * 0.25f; if (wy <= 0.f) continue;
        for (int ix = 0; ix < 9; ++ix) {
            int x = xs + ix; if (x < 0 || x >= WOp) continue;
            float wx = 1.f - fabsf((float)x - cx) * 0.25f; if (wx <= 0.f) continue;
            float w = wy * wx;
            acc  += w * hm[((size_t)b * HO + y) * WOp + x];
            wsum += w;
        }
    }
    resized[o] = acc / wsum;
}

// ---------- pack conv weights into CDNA5 16-bit A-matrix lane layout ----------
__device__ __forceinline__ int a_frag_k(int lane, int e) {
    return ((lane >= 16) ? 16 : 0) + e;
}

__global__ void k_pack_w1(const float* __restrict__ cw1, __bf16* __restrict__ pw1) {
    int idx = blockIdx.x * blockDim.x + threadIdx.x;
    if (idx >= 18 * 2 * 32 * 16) return;
    int e = idx & 15, lane = (idx >> 4) & 31, mt = (idx >> 9) & 1, t = idx >> 10;
    int pos = t >> 1, half = t & 1;
    int ky = pos / 3, kx = pos % 3;
    int ic = 32 * half + a_frag_k(lane, e);
    int oc = mt * 16 + (lane & 15);
    pw1[idx] = (__bf16)cw1[((oc * 64 + ic) * 3 + ky) * 3 + kx];
}

__global__ void k_pack_w2(const float* __restrict__ cw2, __bf16* __restrict__ pw2) {
    int idx = blockIdx.x * blockDim.x + threadIdx.x;
    if (idx >= 9 * 32 * 16) return;
    int e = idx & 15, lane = (idx >> 4) & 31, t = idx >> 9;
    int ky = t / 3, kx = t % 3;
    int ic = a_frag_k(lane, e);
    int oc = lane & 15;
    pw2[idx] = (__bf16)cw2[((oc * 32 + ic) * 3 + ky) * 3 + kx];
}

// ---------- conv1: 64ch -> 32ch 3x3 + ReLU, implicit GEMM via WMMA bf16 ----------
// Weights pulled into LDS by the Tensor Data Mover (wave 0 only, TENSORcnt-drained).
// Output layout is channel-innermost [b][y][x][32] so each lane's 8-channel
// epilogue slice is one contiguous 16-byte store, and conv2 can async-copy it.
__global__ __launch_bounds__(256) void k_conv1(const float* __restrict__ sam,
                                               const float* __restrict__ resized,
                                               const __bf16* __restrict__ pw1,
                                               const float* __restrict__ cb1,
                                               __bf16* __restrict__ c1out) {
    __shared__ __align__(32) __bf16 tile[3 * 66 * 64];
    __shared__ __align__(64) __bf16 wlds[18 * 2 * 32 * 16];
    int b = blockIdx.z, y = blockIdx.y, xbase = blockIdx.x * 64;
    int tid = threadIdx.x;

    if (tid < 32)  // one TDM issue per block; TENSORcnt tracks completion
        tdm_load_1d((unsigned)(uintptr_t)wlds, pw1, 18 * 2 * 32 * 16 * 2);

    if (tid < 198) {  // 3 rows * 66 cols; fused resize-product, channel-innermost
        int r = tid / 66, col = tid % 66;
        int gy = y + r - 1, gx = xbase + col - 1;
        bool inb = (gy >= 0) && (gy < 256) && (gx >= 0) && (gx < 256);
        float rv = inb ? resized[((size_t)b * 256 + gy) * 256 + gx] : 0.f;
        for (int ch = 0; ch < 32; ++ch) {
            float s = inb ? sam[(((size_t)b * 32 + ch) * 256 + gy) * 256 + gx] : 0.f;
            tile[(r * 66 + col) * 64 + ch]      = (__bf16)s;
            tile[(r * 66 + col) * 64 + 32 + ch] = (__bf16)(rv * s);
        }
    }
    __builtin_amdgcn_s_wait_tensorcnt(0);  // no-op for waves that issued nothing
    __syncthreads();

    int wave = tid >> 5, lane = tid & 31;
    int mt = wave & 1, nt = wave >> 1;       // 2 M-tiles x 4 N-tiles
    const v16bf* Aw = (const v16bf*)wlds;
    const v16bf* Bp = (const v16bf*)tile;
    v8f acc = {0.f, 0.f, 0.f, 0.f, 0.f, 0.f, 0.f, 0.f};
    for (int t = 0; t < 18; ++t) {           // K = 576 in chunks of 32
        int pos = t >> 1, half = t & 1;
        int ky = pos / 3, kx = pos % 3;
        v16bf a = Aw[(t * 2 + mt) * 32 + lane];
        int col = nt * 16 + (lane & 15) + kx;
        v16bf bb = Bp[(ky * 66 + col) * 4 + half * 2 + (lane >> 4)];
        acc = __builtin_amdgcn_wmma_f32_16x16x32_bf16(false, a, false, bb,
                                                      (short)0, acc, false, false);
    }
    int px  = xbase + nt * 16 + (lane & 15);
    int ocb = mt * 16 + ((lane >= 16) ? 8 : 0);
    v8bf outv;
    for (int r = 0; r < 8; ++r) {
        float v = acc[r] + cb1[ocb + r];
        outv[r] = (__bf16)(v > 0.f ? v : 0.f);
    }
    *(v8bf*)(c1out + ((size_t)b * 65536 + y * 256 + px) * 32 + ocb) = outv;
}

// ---------- conv2: 32ch -> 16ch 3x3 + ReLU ----------
// Input staging uses per-lane GLOBAL_LOAD_ASYNC_TO_LDS_B128 (ASYNCcnt): a pixel's
// 32 channels are 64 contiguous bytes in both global and the LDS tile, so the
// copy needs no shuffle. Weights come in via TDM.
__global__ __launch_bounds__(256) void k_conv2(const __bf16* __restrict__ c1,
                                               const __bf16* __restrict__ pw2,
                                               const float* __restrict__ cb2,
                                               __bf16* __restrict__ c2out) {
    __shared__ __align__(32) __bf16 tile[3 * 130 * 32];
    __shared__ __align__(64) __bf16 wlds[9 * 32 * 16];
    int b = blockIdx.z, y = blockIdx.y, xbase = blockIdx.x * 128;
    int tid = threadIdx.x;

    if (tid < 32)
        tdm_load_1d((unsigned)(uintptr_t)wlds, pw2, 9 * 32 * 16 * 2);

    unsigned tile_base = (unsigned)(uintptr_t)tile;
    for (int posn = tid; posn < 3 * 130; posn += 256) {
        int r = posn / 130, col = posn % 130;
        int gy = y + r - 1, gx = xbase + col - 1;
        bool inb = (gy >= 0) && (gy < 256) && (gx >= 0) && (gx < 256);
        unsigned ldsoff = tile_base + (unsigned)((r * 130 + col) * 32) * 2u;
        if (inb) {
            unsigned long long ga = (unsigned long long)(uintptr_t)
                (c1 + ((size_t)b * 65536 + gy * 256 + gx) * 32);
            asm volatile(
                "global_load_async_to_lds_b128 %0, %1, off\n\t"
                "global_load_async_to_lds_b128 %0, %1, off offset:16\n\t"
                "global_load_async_to_lds_b128 %0, %1, off offset:32\n\t"
                "global_load_async_to_lds_b128 %0, %1, off offset:48"
                :: "v"(ldsoff), "v"(ga) : "memory");
        } else {
            v8bf z = {};
            *(v8bf*)(tile + (r * 130 + col) * 32)      = z;
            *(v8bf*)(tile + (r * 130 + col) * 32 + 8)  = z;
            *(v8bf*)(tile + (r * 130 + col) * 32 + 16) = z;
            *(v8bf*)(tile + (r * 130 + col) * 32 + 24) = z;
        }
    }
    asm volatile("s_wait_asynccnt 0x0" ::: "memory");
    __builtin_amdgcn_s_wait_tensorcnt(0);
    __syncthreads();

    int wave = tid >> 5, lane = tid & 31;
    int nt = wave;                           // 1 M-tile x 8 N-tiles (128 px)
    const v16bf* Aw = (const v16bf*)wlds;
    const v16bf* Bp = (const v16bf*)tile;
    v8f acc = {0.f, 0.f, 0.f, 0.f, 0.f, 0.f, 0.f, 0.f};
    for (int t = 0; t < 9; ++t) {            // K = 288 in chunks of 32
        int ky = t / 3, kx = t % 3;
        v16bf a = Aw[t * 32 + lane];
        int col = nt * 16 + (lane & 15) + kx;
        v16bf bb = Bp[(ky * 130 + col) * 2 + (lane >> 4)];
        acc = __builtin_amdgcn_wmma_f32_16x16x32_bf16(false, a, false, bb,
                                                      (short)0, acc, false, false);
    }
    int px  = xbase + nt * 16 + (lane & 15);
    int ocb = (lane >= 16) ? 8 : 0;
    v8bf outv;
    for (int r = 0; r < 8; ++r) {
        float v = acc[r] + cb2[ocb + r];
        outv[r] = (__bf16)(v > 0.f ? v : 0.f);
    }
    *(v8bf*)(c2out + ((size_t)b * 65536 + y * 256 + px) * 16 + ocb) = outv;
}

// ---------- conv3: 1x1 16ch -> 1 + sigmoid (channel-innermost: 2x16B loads) ------
__global__ void k_conv3(const __bf16* __restrict__ c2, const float* __restrict__ cw3,
                        const float* __restrict__ cb3, float* __restrict__ refined) {
    int o = blockIdx.x * blockDim.x + threadIdx.x;
    if (o >= BATCH * HS * WSp) return;
    const v8bf* p = (const v8bf*)(c2 + (size_t)o * 16);
    v8bf a = p[0], bq = p[1];
    float acc = cb3[0];
    for (int ch = 0; ch < 8; ++ch) {
        acc += cw3[ch]     * (float)a[ch];
        acc += cw3[ch + 8] * (float)bq[ch];
    }
    refined[o] = 1.f / (1.f + expf(-acc));
}

// ---------- bilinear upsample 256 -> 1024 (half-pixel centers, edge clamp) ----------
__global__ void k_upsample(const float* __restrict__ refined, float* __restrict__ out) {
    int o = blockIdx.x * blockDim.x + threadIdx.x;
    if (o >= BATCH * HO * WOp) return;
    int b = o >> 20; int rem = o & 0xFFFFF;
    int Y = rem >> 10, X = rem & 1023;
    float u = (X + 0.5f) * 0.25f - 0.5f;
    float v = (Y + 0.5f) * 0.25f - 0.5f;
    int x0 = (int)floorf(u), y0 = (int)floorf(v);
    float fx = u - (float)x0, fy = v - (float)y0;
    int x0c = min(max(x0, 0), 255), x1c = min(max(x0 + 1, 0), 255);
    int y0c = min(max(y0, 0), 255), y1c = min(max(y0 + 1, 0), 255);
    const float* rp = refined + (size_t)b * 65536;
    float v00 = rp[y0c * 256 + x0c], v01 = rp[y0c * 256 + x1c];
    float v10 = rp[y1c * 256 + x0c], v11 = rp[y1c * 256 + x1c];
    out[o] = (1.f - fy) * ((1.f - fx) * v00 + fx * v01)
           +        fy  * ((1.f - fx) * v10 + fx * v11);
}

// ---------- per-batch: top-20 -> NMS -> depth MLP -> poses / scores ----------
__global__ __launch_bounds__(256) void k_candidates(
        const float* __restrict__ placement, const float* __restrict__ scene,
        const float* __restrict__ mw1, const float* __restrict__ mb1,
        const float* __restrict__ mw2, const float* __restrict__ mb2,
        const float* __restrict__ mw3, const float* __restrict__ mb3,
        float* __restrict__ out_cand, float* __restrict__ out_poses,
        float* __restrict__ out_scores) {
    __shared__ float rv[256]; __shared__ int ri[256];
    __shared__ int   topi[TOPK];
    __shared__ float coords[TOPK * 2];
    __shared__ int   sel[NUM_CAND];
    __shared__ float scene_s[SCENE_DIM];
    __shared__ float h1[128]; __shared__ float h2[64];

    int b = blockIdx.x, tid = threadIdx.x;
    const float* pm = placement + (size_t)b * HO * WOp;
    for (int i = tid; i < SCENE_DIM; i += 256) scene_s[i] = scene[b * SCENE_DIM + i];

    // iterative argmax under strict key (value desc, index asc) = lax.top_k order
    float pv = 3.4e38f; int pi = -1;
    for (int k = 0; k < TOPK; ++k) {
        float bv = -3.4e38f; int bi = 0x7fffffff;
        for (int i = tid; i < HO * WOp; i += 256) {
            float v = pm[i];
            bool ok = (v < pv) || (v == pv && i > pi);
            if (ok && (v > bv || (v == bv && i < bi))) { bv = v; bi = i; }
        }
        rv[tid] = bv; ri[tid] = bi; __syncthreads();
        for (int s = 128; s > 0; s >>= 1) {
            if (tid < s) {
                float v2 = rv[tid + s]; int i2 = ri[tid + s];
                if (v2 > rv[tid] || (v2 == rv[tid] && i2 < ri[tid])) { rv[tid] = v2; ri[tid] = i2; }
            }
            __syncthreads();
        }
        if (tid == 0) topi[k] = ri[0];
        __syncthreads();
        pv = rv[0]; pi = ri[0];
        __syncthreads();
    }

    if (tid == 0) {
        for (int k = 0; k < TOPK; ++k) {
            coords[2 * k]     = (float)(topi[k] % WOp) / 1023.f;
            coords[2 * k + 1] = (float)(topi[k] / WOp) / 1023.f;
        }
        bool mask[TOPK]; int cnt = 0;
        for (int i = 0; i < TOPK; ++i) {
            bool close = false;
            for (int j = 0; j < i; ++j) {
                if (mask[j]) {
                    float dx = coords[2 * i] - coords[2 * j];
                    float dy = coords[2 * i + 1] - coords[2 * j + 1];
                    if (sqrtf(dx * dx + dy * dy) < 0.1f) close = true;
                }
            }
            bool keep = (!close) && (cnt < NUM_CAND);
            mask[i] = keep; cnt += keep ? 1 : 0;
        }
        int n = 0;
        for (int i = 0; i < TOPK && n < NUM_CAND; ++i) if (mask[i])  sel[n++] = i;
        for (int i = 0; i < TOPK && n < NUM_CAND; ++i) if (!mask[i]) sel[n++] = i;
    }
    __syncthreads();

    for (int c = 0; c < NUM_CAND; ++c) {
        float cx = coords[2 * sel[c]], cy = coords[2 * sel[c] + 1];
        if (tid < 128) {
            float acc = mb1[tid] + cx * mw1[tid] + cy * mw1[128 + tid];
            for (int i = 0; i < SCENE_DIM; ++i) acc += scene_s[i] * mw1[(2 + i) * 128 + tid];
            h1[tid] = acc > 0.f ? acc : 0.f;
        }
        __syncthreads();
        if (tid < 64) {
            float acc = mb2[tid];
            for (int i = 0; i < 128; ++i) acc += h1[i] * mw2[i * 64 + tid];
            h2[tid] = acc > 0.f ? acc : 0.f;
        }
        __syncthreads();
        if (tid == 0) {
            float acc = mb3[0];
            for (int i = 0; i < 64; ++i) acc += h2[i] * mw3[i];
            float depth = 1.f / (1.f + expf(-acc));
            out_cand[(b * NUM_CAND + c) * 2 + 0] = cx;
            out_cand[(b * NUM_CAND + c) * 2 + 1] = cy;
            float* pp = out_poses + (b * NUM_CAND + c) * 7;
            pp[0] = cx * 2.f - 1.f; pp[1] = cy * 2.f - 1.f; pp[2] = depth * 2.f - 1.f;
            pp[3] = 0.f; pp[4] = 0.f; pp[5] = 0.f; pp[6] = 1.f;
            int px = min(max((int)(cx * 1023.f), 0), 1023);
            int py = min(max((int)(cy * 1023.f), 0), 1023);
            out_scores[b * NUM_CAND + c] = pm[py * WOp + px];
        }
        __syncthreads();
    }
}

extern "C" void kernel_launch(void* const* d_in, const int* in_sizes, int n_in,
                              void* d_out, int out_size, void* d_ws, size_t ws_size,
                              hipStream_t stream) {
    const float* hm    = (const float*)d_in[0];
    const float* sam   = (const float*)d_in[1];
    const float* scene = (const float*)d_in[2];
    const float* cw1 = (const float*)d_in[3];  const float* cb1 = (const float*)d_in[4];
    const float* cw2 = (const float*)d_in[5];  const float* cb2 = (const float*)d_in[6];
    const float* cw3 = (const float*)d_in[7];  const float* cb3 = (const float*)d_in[8];
    const float* mw1 = (const float*)d_in[9];  const float* mb1 = (const float*)d_in[10];
    const float* mw2 = (const float*)d_in[11]; const float* mb2 = (const float*)d_in[12];
    const float* mw3 = (const float*)d_in[13]; const float* mb3 = (const float*)d_in[14];

    char* ws = (char*)d_ws;
    float*  resized = (float*)(ws + OFF_RESIZED);
    __bf16* pw1     = (__bf16*)(ws + OFF_PW1);
    __bf16* pw2     = (__bf16*)(ws + OFF_PW2);
    __bf16* c1      = (__bf16*)(ws + OFF_C1);
    __bf16* c2      = (__bf16*)(ws + OFF_C2);
    float*  refined = (float*)(ws + OFF_REF);

    float* out        = (float*)d_out;
    float* placement  = out;                       // 16*1024*1024
    float* out_cand   = out + 16777216;            // 16*5*2
    float* out_poses  = out + 16777376;            // 16*5*7
    float* out_scores = out + 16777936;            // 16*5

    k_downsample<<<(BATCH * HS * WSp + 255) / 256, 256, 0, stream>>>(hm, resized);
    k_pack_w1<<<(18432 + 255) / 256, 256, 0, stream>>>(cw1, pw1);
    k_pack_w2<<<(4608 + 255) / 256, 256, 0, stream>>>(cw2, pw2);
    k_conv1<<<dim3(4, 256, BATCH), 256, 0, stream>>>(sam, resized, pw1, cb1, c1);
    k_conv2<<<dim3(2, 256, BATCH), 256, 0, stream>>>(c1, pw2, cb2, c2);
    k_conv3<<<(BATCH * HS * WSp + 255) / 256, 256, 0, stream>>>(c2, cw3, cb3, refined);
    k_upsample<<<(BATCH * HO * WOp) / 256, 256, 0, stream>>>(refined, placement);
    k_candidates<<<BATCH, 256, 0, stream>>>(placement, scene, mw1, mb1, mw2, mb2,
                                            mw3, mb3, out_cand, out_poses, out_scores);
}
